// NNUENetworkSparseCOO_64158221467786
// MI455X (gfx1250) — matrix operand order repaired and checked by reading, not compiled, not run on hardware
//
#include <hip/hip_runtime.h>

typedef __attribute__((ext_vector_type(16))) _Float16 v16h;
typedef __attribute__((ext_vector_type(8)))  float    v8f;
typedef __attribute__((ext_vector_type(4)))  float    f4;
typedef __attribute__((ext_vector_type(4)))  _Float16 h4;

#define BATCH          32768
#define HIDDEN         256
#define ACTIVE         32
#define POS_PER_BLOCK  64
#define ACC_STRIDE     520   // 512 + 8 halfs pad -> rows land 4 banks apart
#define X1_STRIDE      34    // 32 + 2 halfs pad  -> rows land 17 banks apart

union Frag { v16h h; unsigned int u[8]; };

// ---------------------------------------------------------------------------
// Tiny kernel: convert L1/L2 weights to f16 once per launch into workspace.
// ---------------------------------------------------------------------------
__global__ void nnue_cvt_weights(const float* __restrict__ l1w,
                                 const float* __restrict__ l2w,
                                 _Float16* __restrict__ l1h,
                                 _Float16* __restrict__ l2h) {
  int t = blockIdx.x * blockDim.x + threadIdx.x;
  int stride = gridDim.x * blockDim.x;
  for (int i = t; i < 32 * 512; i += stride) l1h[i] = (_Float16)l1w[i];
  for (int i = t; i < 32 * 32;  i += stride) l2h[i] = (_Float16)l2w[i];
}

// ---------------------------------------------------------------------------
// Fused NNUE forward: float4 gather-sum FT -> LDS, WMMA L1/L2, shfl L3.
// ---------------------------------------------------------------------------
__global__ __launch_bounds__(256) void nnue_forward(
    const int*   __restrict__ wIdx,  const int*   __restrict__ bIdx,
    const float* __restrict__ stm,   const float* __restrict__ ftW,
    const float* __restrict__ ftB,
    const _Float16* __restrict__ l1h, const float* __restrict__ l1b,
    const _Float16* __restrict__ l2h, const float* __restrict__ l2b,
    const float* __restrict__ l3w,   const float* __restrict__ l3b,
    float* __restrict__ out)
{
  __shared__ alignas(16) _Float16 accS[POS_PER_BLOCK * ACC_STRIDE]; // 66,560 B
  __shared__ alignas(16) _Float16 x1S [POS_PER_BLOCK * X1_STRIDE];  //  4,352 B
  __shared__ float partS[POS_PER_BLOCK][2];                         //    512 B

  const int t      = threadIdx.x;
  const int block0 = blockIdx.x * POS_PER_BLOCK;

  // -------- Phase 1: feature transform (float4 gather-sum, clip, stm mix) --
  {
    const int col4 = t & 63;          // group of 4 hidden columns
    const int pg   = t >> 6;          // position sub-slot 0..3 (wave-uniform)
    const f4* __restrict__ ftW4 = (const f4*)ftW;
    const f4 bias = ((const f4*)ftB)[col4];

    for (int p4 = 0; p4 < POS_PER_BLOCK / 4; ++p4) {
      const int p  = p4 * 4 + pg;
      const int gp = block0 + p;
      const int* __restrict__ wi = wIdx + gp * ACTIVE;   // wave-uniform
      const int* __restrict__ bi = bIdx + gp * ACTIVE;
      f4 ws = bias, bs = bias;
      #pragma unroll 8
      for (int a = 0; a < ACTIVE; ++a) {
        ws += ftW4[wi[a] * (HIDDEN / 4) + col4];   // global_load_b128 gather
        bs += ftW4[bi[a] * (HIDDEN / 4) + col4];
      }
      const float s = stm[gp];
      h4 h0, h1;
      #pragma unroll
      for (int c = 0; c < 4; ++c) {
        const float w = fminf(fmaxf(ws[c], 0.f), 1.f);
        const float b = fminf(fmaxf(bs[c], 0.f), 1.f);
        h0[c] = (_Float16)(s * w + (1.f - s) * b);   // first 256 cols
        h1[c] = (_Float16)(s * b + (1.f - s) * w);   // last  256 cols
      }
      *(h4*)(&accS[p * ACC_STRIDE + col4 * 4])       = h0;   // 8B LDS stores
      *(h4*)(&accS[p * ACC_STRIDE + 256 + col4 * 4]) = h1;
    }
  }
  __syncthreads();

  // -------- Phase 2: WMMA layers. 8 waves = 4 batch tiles x 2 n-tiles -----
  const int wave  = t >> 5;
  const int lane  = t & 31;
  const int tile  = wave >> 1;          // 0..3 : 16-row batch tile
  const int ntile = wave & 1;           // 0..1 : output columns [0:16) / [16:32)
  const int mrow  = lane & 15;
  const int hi    = lane >> 4;          // which 16-lane half
  const int row0  = tile * 16;
  const int ncol  = ntile * 16 + mrow;  // this lane's output-neuron column

  // K offsets inside one 32-wide step, per documented 16-bit fragment maps.
  int kA[8], kB[8];
  #pragma unroll
  for (int v = 0; v < 8; ++v) {
    kA[v] = ((v >= 4) ? 16 : 0) + (v & 3) * 2 + hi * 8;  // A 16x32 map
    kB[v] = ((v >= 4) ? 16 : 0) + (v & 3) * 2 + hi * 8;  // B 32x16 map
  }

  // ---- L1: [16x512] @ [512x32] via 16 K-steps, 2 interleaved accumulators
  v8f cA = {}, cB = {};
  const _Float16* __restrict__ l1row = l1h + ncol * 512;
  #pragma unroll 4
  for (int ks = 0; ks < 16; ++ks) {
    const int kbase = ks * 32;
    Frag a, b;
    #pragma unroll
    for (int v = 0; v < 8; ++v) {
      a.u[v] = *(const unsigned int*)(&accS[(row0 + mrow) * ACC_STRIDE + kbase + kA[v]]);
      b.u[v] = *(const unsigned int*)(&l1row[kbase + kB[v]]);
    }
    if (ks & 1)
      cB = __builtin_amdgcn_wmma_f32_16x16x32_f16(false, a.h, false, b.h, (short)0, cB, false, false);
    else
      cA = __builtin_amdgcn_wmma_f32_16x16x32_f16(false, a.h, false, b.h, (short)0, cA, false, false);
  }

  // x1 = clip(L1 + bias) -> staged to LDS for the L2 A-fragment
  {
    const float b1 = l1b[ncol];
    #pragma unroll
    for (int r = 0; r < 8; ++r) {
      const int m = r + hi * 8;                    // C/D layout: M=r / M=r+8
      float v = cA[r] + cB[r] + b1;
      v = fminf(fmaxf(v, 0.f), 1.f);
      x1S[(row0 + m) * X1_STRIDE + ncol] = (_Float16)v;
    }
  }
  __syncthreads();

  // ---- L2: [16x32] @ [32x32], single WMMA per n-tile ----
  v8f c2 = {};
  {
    Frag a, b;
    const _Float16* __restrict__ l2row = l2h + ncol * 32;
    #pragma unroll
    for (int v = 0; v < 8; ++v) {
      a.u[v] = *(const unsigned int*)(&x1S[(row0 + mrow) * X1_STRIDE + kA[v]]);
      b.u[v] = *(const unsigned int*)(&l2row[kB[v]]);
    }
    c2 = __builtin_amdgcn_wmma_f32_16x16x32_f16(false, a.h, false, b.h, (short)0, c2, false, false);
  }

  // ---- L3: x2 @ l3_w^T via butterfly reduce within each 16-lane half ----
  {
    const float b2 = l2b[ncol];
    const float w3 = l3w[ncol];
    float rs[8];
    #pragma unroll
    for (int r = 0; r < 8; ++r) {
      float v = c2[r] + b2;
      v = fminf(fmaxf(v, 0.f), 1.f);
      rs[r] = v * w3;
    }
    #pragma unroll
    for (int r = 0; r < 8; ++r) {
      float v = rs[r];
      v += __shfl_xor(v, 1, 32);
      v += __shfl_xor(v, 2, 32);
      v += __shfl_xor(v, 4, 32);
      v += __shfl_xor(v, 8, 32);   // stays within 16-lane half
      rs[r] = v;
    }
    if (mrow == 0) {
      #pragma unroll
      for (int r = 0; r < 8; ++r)
        partS[row0 + r + hi * 8][ntile] = rs[r];
    }
  }
  __syncthreads();

  if (t < POS_PER_BLOCK)
    out[block0 + t] = partS[t][0] + partS[t][1] + l3b[0];
}

// ---------------------------------------------------------------------------
extern "C" void kernel_launch(void* const* d_in, const int* in_sizes, int n_in,
                              void* d_out, int out_size, void* d_ws, size_t ws_size,
                              hipStream_t stream) {
  const int*   wIdx = (const int*)  d_in[0];
  const int*   bIdx = (const int*)  d_in[1];
  const float* stm  = (const float*)d_in[2];
  const float* ftW  = (const float*)d_in[3];
  const float* ftB  = (const float*)d_in[4];
  const float* l1w  = (const float*)d_in[5];
  const float* l1b  = (const float*)d_in[6];
  const float* l2w  = (const float*)d_in[7];
  const float* l2b  = (const float*)d_in[8];
  const float* l3w  = (const float*)d_in[9];
  const float* l3b  = (const float*)d_in[10];
  float* out = (float*)d_out;

  _Float16* l1h = (_Float16*)d_ws;            // 32*512 halfs
  _Float16* l2h = l1h + 32 * 512;             // 32*32  halfs  (34,816 B total)

  nnue_cvt_weights<<<68, 256, 0, stream>>>(l1w, l2w, l1h, l2h);
  nnue_forward<<<BATCH / POS_PER_BLOCK, 256, 0, stream>>>(
      wIdx, bIdx, stm, ftW, ftB, l1h, l1b, l2h, l2b, l3w, l3b, out);
}
